// FC_MT_LSTM_37125697307357
// MI455X (gfx1250) — matrix operand order repaired and checked by reading, not compiled, not run on hardware
//
#include <hip/hip_runtime.h>
#include <hip/hip_bf16.h>

// Problem constants from the reference
#define BB   2048
#define SS   256
#define TTT  64
#define FTF  32
#define HH   128
#define EE   256
#define EPSL 0.001f

typedef __attribute__((ext_vector_type(16))) _Float16 v16h;
typedef __attribute__((ext_vector_type(8)))  _Float16 v8h;
typedef __attribute__((ext_vector_type(8)))  float    v8f;

// ---------------------------------------------------------------------------
// WMMA fragment loaders (CDNA5 v_wmma_f32_16x16x32_f16 layouts, wave32)
// A (16x32, f16): lanes 0-15 hold M=lane, K = [0..7]+[16..23]; lanes 16-31
//                 hold M=lane-16, K = [8..15]+[24..31].
// B (32x16, f16): lane holds column N=lane&15; lanes 0-15 K=[0..7]+[16..23],
//                 lanes 16-31 K=[8..15]+[24..31] (column-major source).
// C/D (16x16,f32): d[r] -> M = r + 8*(lane>=16), N = lane&15.
// ---------------------------------------------------------------------------
__device__ __forceinline__ v16h load_fragA(const _Float16* __restrict__ row, int k0, int lane) {
  int base = k0 + ((lane & 16) ? 8 : 0);
  v8h lo = *(const v8h*)(row + base);
  v8h hi = *(const v8h*)(row + base + 16);
  v16h r;
#pragma unroll
  for (int i = 0; i < 8; ++i) { r[i] = lo[i]; r[i + 8] = hi[i]; }
  return r;
}
__device__ __forceinline__ v16h load_fragB(const _Float16* __restrict__ col, int k0, int lane) {
  int base = k0 + ((lane & 16) ? 8 : 0);
  v8h lo = *(const v8h*)(col + base);
  v8h hi = *(const v8h*)(col + base + 16);
  v16h r;
#pragma unroll
  for (int i = 0; i < 8; ++i) { r[i] = lo[i]; r[i + 8] = hi[i]; }
  return r;
}
#define WMMA_F16(A, Bf, C) \
  __builtin_amdgcn_wmma_f32_16x16x32_f16(false, (A), false, (Bf), (short)0, (C), false, false)

// Branch-free transcendentals on the raw hardware v_exp_f32 / v_rcp_f32 path.
// (~1 ulp rcp error is far below the f16 storage precision of h.)
__device__ __forceinline__ float sigm(float x) {
  return __builtin_amdgcn_rcpf(1.0f + __expf(-x));
}
__device__ __forceinline__ float tanh_fast(float x) {
  float xc = fminf(fmaxf(x, -15.0f), 15.0f);       // keep e^{2x} finite
  return 1.0f - 2.0f * __builtin_amdgcn_rcpf(__expf(2.0f * xc) + 1.0f);
}

// ---------------------------------------------------------------------------
// Prep: f32 [R][C] -> f16 [C][R] (column-major) so B-fragments load contiguously
// ---------------------------------------------------------------------------
__global__ void tr_f32_to_f16_kernel(const float* __restrict__ src,
                                     _Float16* __restrict__ dst, int R, int C) {
  int idx = blockIdx.x * 256 + threadIdx.x;
  if (idx >= R * C) return;
  int r = idx / C, c = idx % C;
  dst[(size_t)c * R + r] = (_Float16)src[idx];
}

// ---------------------------------------------------------------------------
// Spatial path: conv1(3,1,32)+bn+relu -> conv2(3,32,64)+bn+relu -> max over S
//               -> FC(64->64)+relu.  One block per batch row, 256 threads.
// ---------------------------------------------------------------------------
__global__ void spatial_kernel(const float* __restrict__ sp,
                               const float* __restrict__ w1, const float* __restrict__ b1,
                               const float* __restrict__ g1, const float* __restrict__ bb1,
                               const float* __restrict__ w2, const float* __restrict__ b2,
                               const float* __restrict__ g2, const float* __restrict__ bb2,
                               const float* __restrict__ spw, const float* __restrict__ spb,
                               float* __restrict__ feat) {
  __shared__ float x0[SS + 2];
  __shared__ float x1[(SS + 2) * 32];
  __shared__ float wred[8 * 64];
  __shared__ float maxv[64];
  int b = blockIdx.x, tid = threadIdx.x;
  const float inv = rsqrtf(1.0f + EPSL);

  x0[tid + 1] = sp[(size_t)b * SS + tid];
  if (tid == 0) { x0[0] = 0.0f; x0[SS + 1] = 0.0f; }
  __syncthreads();

  float xm = x0[tid], xc = x0[tid + 1], xp = x0[tid + 2];
#pragma unroll
  for (int c = 0; c < 32; ++c) {
    float y = xm * w1[c] + xc * w1[32 + c] + xp * w1[64 + c] + b1[c];
    y = y * (g1[c] * inv) + bb1[c];
    x1[(tid + 1) * 32 + c] = fmaxf(y, 0.0f);
  }
  if (tid < 32) { x1[tid] = 0.0f; x1[(SS + 1) * 32 + tid] = 0.0f; }
  __syncthreads();

  float acc[64];
#pragma unroll
  for (int c = 0; c < 64; ++c) acc[c] = b2[c];
  for (int k = 0; k < 3; ++k)
    for (int ci = 0; ci < 32; ++ci) {
      float xv = x1[(tid + k) * 32 + ci];
      const float* wrow = w2 + (k * 32 + ci) * 64;
#pragma unroll
      for (int c = 0; c < 64; ++c) acc[c] += xv * wrow[c];
    }

  int lane = tid & 31, wid = tid >> 5;
  for (int c = 0; c < 64; ++c) {
    float y = fmaxf(acc[c] * (g2[c] * inv) + bb2[c], 0.0f);
#pragma unroll
    for (int off = 16; off > 0; off >>= 1) y = fmaxf(y, __shfl_xor(y, off, 32));
    if (lane == 0) wred[wid * 64 + c] = y;
  }
  __syncthreads();
  if (tid < 64) {
    float m = wred[tid];
#pragma unroll
    for (int w = 1; w < 8; ++w) m = fmaxf(m, wred[w * 64 + tid]);
    maxv[tid] = m;
  }
  __syncthreads();
  if (tid < 64) {
    float y = spb[tid];
    for (int c = 0; c < 64; ++c) y += maxv[c] * spw[c * 64 + tid];
    feat[(size_t)b * 64 + tid] = fmaxf(y, 0.0f);
  }
}

// ---------------------------------------------------------------------------
// LSTM: one block = 16 batch rows, full time loop, state resident.
//   grid (128, 2dirs), 256 threads (8 waves). wave w owns h-columns
//   [w*16, w*16+16): its 4 gate tiles (cols +0/+128/+256/+384).
//   The 20 loop-invariant B-fragments (W: 4, U: 16) are preloaded into
//   registers BEFORE the time loop (160 VGPRs) and stay resident via the
//   gfx1250 VGPR-MSB mechanism (__launch_bounds__(256,1) gives headroom).
//   A-fragments (h) are re-read from LDS per use (ds_load_b128).
//   c-state lives in the wave's f32 accumulator fragment across timesteps.
// ---------------------------------------------------------------------------
__global__ void __launch_bounds__(256, 1)
lstm_kernel(const float* __restrict__ temporal,
            const _Float16* __restrict__ WtF, const _Float16* __restrict__ UtF,
            const float* __restrict__ bF,
            const _Float16* __restrict__ WtB, const _Float16* __restrict__ UtB,
            const float* __restrict__ bB,
            _Float16* __restrict__ lstm_out) {
  __shared__ __align__(16) _Float16 hbuf[2][16][HH];
  __shared__ __align__(16) _Float16 xbuf[16][FTF];
  int tid = threadIdx.x;
  int lane = tid & 31, wid = tid >> 5;
  int dir = blockIdx.y;
  int b0 = blockIdx.x * 16;
  const _Float16* Wt = dir ? WtB : WtF;
  const _Float16* Ut = dir ? UtB : UtF;
  const float* bias = dir ? bB : bF;

  for (int i = tid; i < 16 * HH; i += 256) ((_Float16*)hbuf)[i] = (_Float16)0.0f;

  int mrow = lane & 15;              // A-fragment row within the tile
  int ncol = wid * 16 + (lane & 15); // this wave's h-column for this lane
  int mbase = (lane & 16) ? 8 : 0;

  // ---- Preload all loop-invariant B fragments into registers ----
  v16h bx[4];     // x@W  (K=32), one per gate
  v16h bu[4][4];  // h@U  (K=128 in 4 chunks), per gate
  float bv[4];
#pragma unroll
  for (int gi = 0; gi < 4; ++gi) {
    int col = gi * HH + ncol;
    bv[gi] = bias[col];
    bx[gi] = load_fragB(Wt + (size_t)col * FTF, 0, lane);
#pragma unroll
    for (int kc = 0; kc < 4; ++kc)
      bu[gi][kc] = load_fragB(Ut + (size_t)col * HH, kc * 32, lane);
  }

  v8f cst;
#pragma unroll
  for (int r = 0; r < 8; ++r) cst[r] = 0.0f;

#pragma unroll 1
  for (int t = 0; t < TTT; ++t) {
    int tt = dir ? (TTT - 1 - t) : t;
    __syncthreads();
    for (int i = tid; i < 16 * FTF; i += 256) {
      int r = i >> 5, f = i & 31;
      xbuf[r][f] = (_Float16)temporal[((size_t)(b0 + r) * TTT + tt) * FTF + f];
      if (t + 1 < TTT) {
        int tn = dir ? (TTT - 2 - t) : (t + 1);
        __builtin_prefetch(&temporal[((size_t)(b0 + r) * TTT + tn) * FTF + f], 0, 3);
      }
    }
    __syncthreads();

    const _Float16* hrow = &hbuf[t & 1][mrow][0];
    v16h ax = load_fragA(&xbuf[mrow][0], 0, lane);

    v8f z[4];
#pragma unroll
    for (int gi = 0; gi < 4; ++gi) {
      v8f acc;
#pragma unroll
      for (int r = 0; r < 8; ++r) acc[r] = bv[gi];
      acc = WMMA_F16(ax, bx[gi], acc);
#pragma unroll
      for (int kc = 0; kc < 4; ++kc) {
        v16h ah = load_fragA(hrow, kc * 32, lane);  // cheap LDS re-read
        acc = WMMA_F16(ah, bu[gi][kc], acc);
      }
      z[gi] = acc;
    }

    // gates on the accumulator fragments; c-state stays in registers
#pragma unroll
    for (int r = 0; r < 8; ++r) {
      float ig = sigm(z[0][r]);
      float fg = sigm(z[1][r]);
      float gg = tanh_fast(z[2][r]);
      float og = sigm(z[3][r]);
      float c = fg * cst[r] + ig * gg;
      cst[r] = c;
      float h = og * tanh_fast(c);
      int m = r + mbase;
      _Float16 hv = (_Float16)h;
      hbuf[(t + 1) & 1][m][ncol] = hv;
      lstm_out[(((size_t)(b0 + m) * TTT + tt) * (2 * HH)) + dir * HH + ncol] = hv;
    }
  }
}

// ---------------------------------------------------------------------------
// Attention: scores = tanh(lstm_out @ attn_w + b); softmax over T;
//            context = sum_t lstm_out * attn.  One block per batch row.
// ---------------------------------------------------------------------------
__global__ void attn_kernel(const _Float16* __restrict__ lo,
                            const float* __restrict__ aw, const float* __restrict__ ab,
                            float* __restrict__ attn_out, float* __restrict__ context) {
  __shared__ float sc[TTT];
  __shared__ float avw[TTT];
  int b = blockIdx.x, tid = threadIdx.x;
  const _Float16* row = lo + (size_t)b * TTT * (2 * HH);

  int t = tid >> 2, q = tid & 3;   // 4 lanes cooperate on one 256-dot
  float s = 0.0f;
  for (int j = q * 64; j < q * 64 + 64; ++j) s += (float)row[t * (2 * HH) + j] * aw[j];
  s += __shfl_xor(s, 1, 32);
  s += __shfl_xor(s, 2, 32);
  if (q == 0) sc[t] = tanh_fast(s + ab[0]);
  __syncthreads();

  if (tid < TTT) {
    float m = -1e30f;
    for (int i = 0; i < TTT; ++i) m = fmaxf(m, sc[i]);
    float sum = 0.0f;
    for (int i = 0; i < TTT; ++i) sum += __expf(sc[i] - m);
    float a = __expf(sc[tid] - m) * __builtin_amdgcn_rcpf(sum);
    avw[tid] = a;
    attn_out[(size_t)b * TTT + tid] = a;
  }
  __syncthreads();

  float c = 0.0f;
  for (int k = 0; k < TTT; ++k) c += (float)row[k * (2 * HH) + tid] * avw[k];
  context[(size_t)b * (2 * HH) + tid] = c;
}

// ---------------------------------------------------------------------------
// Encoder: z = [feat(64) | context(256)] (K=320) -> e1(256) bn relu
//          -> e2(256) bn relu. WMMA, 16 batch rows per block, 8 waves x 2 tiles.
// ---------------------------------------------------------------------------
__global__ void encoder_kernel(const float* __restrict__ feat, const float* __restrict__ ctx,
                               const _Float16* __restrict__ e1t, const float* __restrict__ e1b,
                               const float* __restrict__ e1g, const float* __restrict__ e1bb,
                               const _Float16* __restrict__ e2t, const float* __restrict__ e2b,
                               const float* __restrict__ e2g, const float* __restrict__ e2bb,
                               float* __restrict__ enc) {
  __shared__ __align__(16) _Float16 zbuf[16][320];
  __shared__ __align__(16) _Float16 h1buf[16][EE];
  int tid = threadIdx.x, lane = tid & 31, wid = tid >> 5;
  int b0 = blockIdx.x * 16;
  const float inv = rsqrtf(1.0f + EPSL);

  for (int i = tid; i < 16 * 320; i += 256) {
    int r = i / 320, c = i % 320;
    float v = (c < 64) ? feat[(size_t)(b0 + r) * 64 + c]
                       : ctx[(size_t)(b0 + r) * 256 + (c - 64)];
    zbuf[r][c] = (_Float16)v;
  }
  __syncthreads();

  int mrow = lane & 15;
  int mbase = (lane & 16) ? 8 : 0;
#pragma unroll
  for (int ct = 0; ct < 2; ++ct) {
    int n = (wid * 2 + ct) * 16 + (lane & 15);
    v8f acc = {};
#pragma unroll
    for (int kc = 0; kc < 10; ++kc) {
      v16h a = load_fragA(&zbuf[mrow][0], kc * 32, lane);
      v16h bf = load_fragB(e1t + (size_t)n * 320, kc * 32, lane);
      acc = WMMA_F16(a, bf, acc);
    }
    float scl = e1g[n] * inv, bsh = e1bb[n], bia = e1b[n];
#pragma unroll
    for (int r = 0; r < 8; ++r) {
      float y = fmaxf((acc[r] + bia) * scl + bsh, 0.0f);
      h1buf[r + mbase][n] = (_Float16)y;
    }
  }
  __syncthreads();

#pragma unroll
  for (int ct = 0; ct < 2; ++ct) {
    int n = (wid * 2 + ct) * 16 + (lane & 15);
    v8f acc = {};
#pragma unroll
    for (int kc = 0; kc < 8; ++kc) {
      v16h a = load_fragA(&h1buf[mrow][0], kc * 32, lane);
      v16h bf = load_fragB(e2t + (size_t)n * 256, kc * 32, lane);
      acc = WMMA_F16(a, bf, acc);
    }
    float scl = e2g[n] * inv, bsh = e2bb[n], bia = e2b[n];
#pragma unroll
    for (int r = 0; r < 8; ++r) {
      float y = fmaxf((acc[r] + bia) * scl + bsh, 0.0f);
      enc[(size_t)(b0 + r + mbase) * EE + n] = y;
    }
  }
}

// ---------------------------------------------------------------------------
// Grouped decoder: per-row weight-gathered matvecs (256->128->64->1).
// One block (128 threads) per batch row; data-dependent group index.
// ---------------------------------------------------------------------------
__global__ void decoder_kernel(const float* __restrict__ enc, const int* __restrict__ group,
                               const float* __restrict__ d1w, const float* __restrict__ d1b,
                               const float* __restrict__ d1g, const float* __restrict__ d1bb,
                               const float* __restrict__ d2w, const float* __restrict__ d2b,
                               const float* __restrict__ d2g, const float* __restrict__ d2bb,
                               const float* __restrict__ dow, const float* __restrict__ dob,
                               float* __restrict__ preds) {
  __shared__ float es[EE];
  __shared__ float h1s[128];
  __shared__ float h2s[64];
  int b = blockIdx.x, tid = threadIdx.x;
  int g = group[b];
  const float inv = rsqrtf(1.0f + EPSL);

  es[tid] = enc[(size_t)b * EE + tid];
  es[tid + 128] = enc[(size_t)b * EE + tid + 128];
  __syncthreads();

  {
    float y = d1b[g * 128 + tid];
    const float* w = d1w + (size_t)g * 256 * 128 + tid;
    for (int d = 0; d < 256; ++d) y += es[d] * w[(size_t)d * 128];
    h1s[tid] = fmaxf(y * (d1g[g * 128 + tid] * inv) + d1bb[g * 128 + tid], 0.0f);
  }
  __syncthreads();
  if (tid < 64) {
    float y = d2b[g * 64 + tid];
    const float* w = d2w + (size_t)g * 128 * 64 + tid;
    for (int d = 0; d < 128; ++d) y += h1s[d] * w[(size_t)d * 64];
    h2s[tid] = fmaxf(y * (d2g[g * 64 + tid] * inv) + d2bb[g * 64 + tid], 0.0f);
  }
  __syncthreads();
  if (tid == 0) {
    float y = dob[g];
    for (int d = 0; d < 64; ++d) y += h2s[d] * dow[g * 64 + d];
    preds[b] = y;
  }
}

// ---------------------------------------------------------------------------
// Launch
// ---------------------------------------------------------------------------
extern "C" void kernel_launch(void* const* d_in, const int* in_sizes, int n_in,
                              void* d_out, int out_size, void* d_ws, size_t ws_size,
                              hipStream_t stream) {
  const float* spatial  = (const float*)d_in[0];
  const float* temporal = (const float*)d_in[1];
  const int*   group    = (const int*)  d_in[2];
  const float* conv1_w = (const float*)d_in[3];
  const float* conv1_b = (const float*)d_in[4];
  const float* sbn1_g  = (const float*)d_in[5];
  const float* sbn1_b  = (const float*)d_in[6];
  const float* conv2_w = (const float*)d_in[7];
  const float* conv2_b = (const float*)d_in[8];
  const float* sbn2_g  = (const float*)d_in[9];
  const float* sbn2_b  = (const float*)d_in[10];
  const float* sp_w    = (const float*)d_in[11];
  const float* sp_b    = (const float*)d_in[12];
  const float* Wf      = (const float*)d_in[13];
  const float* Uf      = (const float*)d_in[14];
  const float* bf      = (const float*)d_in[15];
  const float* Wb      = (const float*)d_in[16];
  const float* Ub      = (const float*)d_in[17];
  const float* bb      = (const float*)d_in[18];
  const float* attn_w  = (const float*)d_in[19];
  const float* attn_b  = (const float*)d_in[20];
  const float* e1_w    = (const float*)d_in[21];
  const float* e1_b    = (const float*)d_in[22];
  const float* e1_g    = (const float*)d_in[23];
  const float* e1_bb   = (const float*)d_in[24];
  const float* e2_w    = (const float*)d_in[25];
  const float* e2_b    = (const float*)d_in[26];
  const float* e2_g    = (const float*)d_in[27];
  const float* e2_bb   = (const float*)d_in[28];
  const float* d1_w    = (const float*)d_in[29];
  const float* d1_b    = (const float*)d_in[30];
  const float* d1_g    = (const float*)d_in[31];
  const float* d1_bb   = (const float*)d_in[32];
  const float* d2_w    = (const float*)d_in[33];
  const float* d2_b    = (const float*)d_in[34];
  const float* d2_g    = (const float*)d_in[35];
  const float* d2_bb   = (const float*)d_in[36];
  const float* do_w    = (const float*)d_in[37];
  const float* do_b    = (const float*)d_in[38];

  // Workspace layout (f16 transposed weights, lstm_out, f32 activations)
  _Float16* WtF = (_Float16*)d_ws;                 // [512][32]
  _Float16* WtB = WtF + 512 * 32;                  // [512][32]
  _Float16* UtF = WtB + 512 * 32;                  // [512][128]
  _Float16* UtB = UtF + 512 * 128;                 // [512][128]
  _Float16* E1T = UtB + 512 * 128;                 // [256][320]
  _Float16* E2T = E1T + 256 * 320;                 // [256][256]
  _Float16* LO  = E2T + 256 * 256;                 // [B][T][256] f16
  float* feat = (float*)(LO + (size_t)BB * TTT * 256);  // [B][64]
  float* ctx  = feat + (size_t)BB * 64;                 // [B][256]
  float* enc  = ctx + (size_t)BB * 256;                 // [B][256]

  float* preds    = (float*)d_out;        // [B]
  float* attn_out = preds + BB;           // [B][T]

  // 1) Transpose/convert weights to f16 column-major
  tr_f32_to_f16_kernel<<<(32 * 512 + 255) / 256, 256, 0, stream>>>(Wf, WtF, 32, 512);
  tr_f32_to_f16_kernel<<<(32 * 512 + 255) / 256, 256, 0, stream>>>(Wb, WtB, 32, 512);
  tr_f32_to_f16_kernel<<<(128 * 512 + 255) / 256, 256, 0, stream>>>(Uf, UtF, 128, 512);
  tr_f32_to_f16_kernel<<<(128 * 512 + 255) / 256, 256, 0, stream>>>(Ub, UtB, 128, 512);
  tr_f32_to_f16_kernel<<<(320 * 256 + 255) / 256, 256, 0, stream>>>(e1_w, E1T, 320, 256);
  tr_f32_to_f16_kernel<<<(256 * 256 + 255) / 256, 256, 0, stream>>>(e2_w, E2T, 256, 256);

  // 2) Spatial conv path
  spatial_kernel<<<BB, 256, 0, stream>>>(spatial, conv1_w, conv1_b, sbn1_g, sbn1_b,
                                         conv2_w, conv2_b, sbn2_g, sbn2_b,
                                         sp_w, sp_b, feat);

  // 3) Bidirectional LSTM (state-resident WMMA, register-resident weights)
  lstm_kernel<<<dim3(BB / 16, 2), 256, 0, stream>>>(temporal, WtF, UtF, bf,
                                                    WtB, UtB, bb, LO);

  // 4) Attention + context (writes attn output)
  attn_kernel<<<BB, 256, 0, stream>>>(LO, attn_w, attn_b, attn_out, ctx);

  // 5) Encoder MLP (WMMA)
  encoder_kernel<<<BB / 16, 256, 0, stream>>>(feat, ctx, E1T, e1_b, e1_g, e1_bb,
                                              E2T, e2_b, e2_g, e2_bb, enc);

  // 6) Grouped decoder heads (writes preds)
  decoder_kernel<<<BB, 128, 0, stream>>>(enc, group, d1_w, d1_b, d1_g, d1_bb,
                                         d2_w, d2_b, d2_g, d2_bb, do_w, do_b, preds);
}
